// MyModel_6305011990587
// MI455X (gfx1250) — matrix operand order repaired and checked by reading, not compiled, not run on hardware
//
#include <hip/hip_runtime.h>

#define NWAY  64
#define KSHOT 16
#define QSHOT 512
#define TOPK  32
#define DIM   768
#define NS    (NWAY * KSHOT)   // 1024 support rows
#define NQ    (NWAY * QSHOT)   // 32768 query rows

#define BM    128              // block tile M
#define BN    128              // block tile N
#define KT    32               // K step (one WMMA K)
#define ABYTES   (BM * KT * 2) // 8 KB per A tile
#define BBYTES   (BN * KT * 2) // 8 KB per B tile
#define BUFBYTES (ABYTES + BBYTES)

typedef __attribute__((ext_vector_type(16))) __bf16 v16bf;
typedef __attribute__((ext_vector_type(8)))  __bf16 v8bf;
typedef __attribute__((ext_vector_type(8)))  float  v8f;

// ---- CDNA5 async global->LDS copy (tracked by ASYNCcnt) ----
__device__ __forceinline__ void async_copy_b128(unsigned lds_off, unsigned long long gaddr) {
    asm volatile("global_load_async_to_lds_b128 %0, %1, off"
                 :: "v"(lds_off), "v"(gaddr) : "memory");
}
__device__ __forceinline__ void wait_async_le4() {
    asm volatile("s_wait_asynccnt 0x4" ::: "memory");
}
__device__ __forceinline__ void wait_async_0() {
    asm volatile("s_wait_asynccnt 0x0" ::: "memory");
}

// ---------------- f32 -> bf16 (round-to-nearest-even) ----------------
__global__ void cvt_bf16_kernel(const float* __restrict__ in,
                                unsigned short* __restrict__ out, int n) {
    int i = blockIdx.x * blockDim.x + threadIdx.x;
    if (i < n) {
        unsigned int u = __float_as_uint(in[i]);
        unsigned int r = u + 0x7FFFu + ((u >> 16) & 1u);
        out[i] = (unsigned short)(r >> 16);
    }
}

// ---------------- exact f32 row norms: one wave32 per row ----------------
__global__ void rownorm_kernel(const float* __restrict__ in,
                               float* __restrict__ out, int nrows) {
    int wave = (blockIdx.x * blockDim.x + threadIdx.x) >> 5;
    int lane = threadIdx.x & 31;
    if (wave >= nrows) return;
    const float* row = in + (size_t)wave * DIM;
    float s = 0.f;
    for (int k = lane; k < DIM; k += 32) { float v = row[k]; s += v * v; }
    for (int off = 16; off; off >>= 1) s += __shfl_down(s, off, 32);
    if (lane == 0) out[wave] = s;
}

// ---------------- distance GEMM: d2 = sn + qn - 2 * s@q^T ----------------
// 128x128 block tile, double-buffered LDS fed by async global->LDS copies,
// 8 waves each computing 32(M) x 64(N) via bf16 WMMA with f32 accumulate.
__global__ __launch_bounds__(256) void dist_gemm_kernel(
    const unsigned short* __restrict__ sbf, const unsigned short* __restrict__ qbf,
    const float* __restrict__ snorm, const float* __restrict__ qnorm,
    float* __restrict__ d2) {
    __shared__ __align__(128) unsigned short ldsbuf[2 * BUFBYTES / 2];

    const int t    = threadIdx.x;
    const int lane = t & 31;
    const int wid  = t >> 5;          // wave 0..7
    const int wm   = wid & 3;         // M quadrant (32 rows each)
    const int wn   = wid >> 2;        // N half (64 cols each)
    const int half = lane >> 4;       // K-half this lane holds
    const int l15  = lane & 15;

    const int m0 = blockIdx.y * BM;
    const int n0 = blockIdx.x * BN;

    const unsigned ldsbase = (unsigned)(size_t)&ldsbuf[0];

    // Per K-step: each thread async-copies 2x16B of A and 2x16B of B.
    auto load_tiles = [&](int buf, int kb) {
#pragma unroll
        for (int i = 0; i < 2; ++i) {
            int c = t + 256 * i;              // 16B chunk id, 512 per tile
            int row = c >> 2;                 // 4 chunks per 64B row
            int sub = c & 3;
            unsigned long long ga = (unsigned long long)(const void*)
                ((const char*)sbf + ((size_t)(m0 + row) * DIM + kb) * 2 + sub * 16);
            async_copy_b128(ldsbase + buf * BUFBYTES + c * 16, ga);
        }
#pragma unroll
        for (int i = 0; i < 2; ++i) {
            int c = t + 256 * i;
            int row = c >> 2;
            int sub = c & 3;
            unsigned long long ga = (unsigned long long)(const void*)
                ((const char*)qbf + ((size_t)(n0 + row) * DIM + kb) * 2 + sub * 16);
            async_copy_b128(ldsbase + buf * BUFBYTES + ABYTES + c * 16, ga);
        }
    };

    v8f acc[2][4];
#pragma unroll
    for (int ms = 0; ms < 2; ++ms)
#pragma unroll
        for (int ns = 0; ns < 4; ++ns) acc[ms][ns] = (v8f){};

    load_tiles(0, 0);                          // prologue

    const int NKB = DIM / KT;                  // 24 K-steps
    for (int kbi = 0; kbi < NKB; ++kbi) {
        int buf = kbi & 1;
        if (kbi + 1 < NKB) {
            load_tiles(buf ^ 1, (kbi + 1) * KT);
            wait_async_le4();                  // current buffer's copies done
        } else {
            wait_async_0();
        }
        __syncthreads();                       // all waves' copies visible

        const __bf16* At = (const __bf16*)((const char*)&ldsbuf[0] + buf * BUFBYTES);
        const __bf16* Bt = (const __bf16*)((const char*)&ldsbuf[0] + buf * BUFBYTES + ABYTES);

        // A fragments: lane l15 holds row M=l15 of the 16x32 subtile, K split by half.
        v16bf afrag[2];
#pragma unroll
        for (int ms = 0; ms < 2; ++ms) {
            int row = wm * 32 + ms * 16 + l15;
            v8bf alo = *(const v8bf*)(At + row * KT + half * 16);
            v8bf ahi = *(const v8bf*)(At + row * KT + half * 16 + 8);
            afrag[ms] = __builtin_shufflevector(alo, ahi,
                                                0, 1, 2, 3, 4, 5, 6, 7,
                                                8, 9, 10, 11, 12, 13, 14, 15);
        }
        // B fragments: lane l15 holds column N=l15; half selects K 0-15 / 16-31.
        v16bf bfrag[4];
#pragma unroll
        for (int ns = 0; ns < 4; ++ns) {
            int row = wn * 64 + ns * 16 + l15;
            bfrag[ns] = *(const v16bf*)(Bt + row * KT + half * 16);
        }
#pragma unroll
        for (int ms = 0; ms < 2; ++ms)
#pragma unroll
            for (int ns = 0; ns < 4; ++ns)
                acc[ms][ns] = __builtin_amdgcn_wmma_f32_16x16x32_bf16(
                    false, afrag[ms], false, bfrag[ns], (short)0, acc[ms][ns],
                    false, false);

        __syncthreads();                       // done reading before next overwrite
    }

    // Epilogue: C/D layout -> M = half*8 + r within subtile, N = l15.
#pragma unroll
    for (int ms = 0; ms < 2; ++ms) {
#pragma unroll
        for (int ns = 0; ns < 4; ++ns) {
            int ng = n0 + wn * 64 + ns * 16 + l15;
            float qn = qnorm[ng];
#pragma unroll
            for (int r = 0; r < 8; ++r) {
                int m = m0 + wm * 32 + ms * 16 + half * 8 + r;
                d2[(size_t)m * NQ + ng] = snorm[m] + qn - 2.0f * acc[ms][ns][r];
            }
        }
    }
}

// ---------------- top-32 smallest per row (sorted ascending) ----------------
__global__ __launch_bounds__(256) void topk_kernel(float* __restrict__ d2,
                                                   float* __restrict__ tval,
                                                   int* __restrict__ tidx) {
    __shared__ float sval[256];
    __shared__ int   sidx[256];
    int row = blockIdx.x;
    float* drow = d2 + (size_t)row * NQ;
    int t = threadIdx.x;
    for (int k = 0; k < TOPK; ++k) {
        float best = 3.4e38f;
        int   bi   = NQ;
        for (int i = t; i < NQ; i += 256) {
            float v = drow[i];
            if (v < best || (v == best && i < bi)) { best = v; bi = i; }
        }
        sval[t] = best; sidx[t] = bi;
        __syncthreads();
        for (int s = 128; s > 0; s >>= 1) {
            if (t < s) {
                float ov = sval[t + s]; int oi = sidx[t + s];
                if (ov < sval[t] || (ov == sval[t] && oi < sidx[t])) {
                    sval[t] = ov; sidx[t] = oi;
                }
            }
            __syncthreads();
        }
        if (t == 0) {
            tval[row * TOPK + k] = sval[0];
            tidx[row * TOPK + k] = sidx[0];
            drow[sidx[0]] = 3.4e38f;   // remove winner for next pass
        }
        __syncthreads();
    }
}

// ---------------- accuracy: single block, no atomics ----------------
__global__ void acc_kernel(const int* __restrict__ tidx, float* __restrict__ out_acc) {
    __shared__ int cnt[256];
    int t = threadIdx.x;
    int c = 0;
    for (int i = t; i < NS * TOPK; i += 256) {
        int cls = (i / TOPK) / KSHOT;
        int idx = tidx[i];
        int lo  = cls * QSHOT;
        c += (idx >= lo && idx < lo + QSHOT) ? 1 : 0;
    }
    cnt[t] = c;
    __syncthreads();
    for (int s = 128; s > 0; s >>= 1) {
        if (t < s) cnt[t] += cnt[t + s];
        __syncthreads();
    }
    if (t == 0) out_acc[0] = (float)cnt[0] / (float)(NS * TOPK);
}

// ---------------- gather retrieved query rows -> sampled_data ----------------
__global__ void gather_kernel(const float* __restrict__ q, const int* __restrict__ tidx,
                              float* __restrict__ sampled) {
    int slot = blockIdx.x;                    // (cls*16+shot)*32 + j
    int src  = tidx[slot];
    const float* srow = q + (size_t)src * DIM;
    float* dst = sampled + (size_t)slot * DIM;
    for (int d = threadIdx.x; d < DIM; d += 256) dst[d] = srow[d];
}

// ---------------- prototype means ----------------
__global__ void proto_kernel(const float* __restrict__ s, const float* __restrict__ sampled,
                             float* __restrict__ proto, float* __restrict__ orig) {
    int cls = blockIdx.x;
    for (int d = threadIdx.x; d < DIM; d += 256) {
        float ssum = 0.f;
        for (int j = 0; j < KSHOT; ++j)
            ssum += s[((size_t)cls * KSHOT + j) * DIM + d];
        float qsum = 0.f;
        const float* sp = sampled + (size_t)cls * (KSHOT * TOPK) * DIM + d;
        for (int j = 0; j < KSHOT * TOPK; ++j) qsum += sp[(size_t)j * DIM];
        proto[(size_t)cls * DIM + d] = (ssum + qsum) / (float)(KSHOT + KSHOT * TOPK);
        orig[(size_t)cls * DIM + d]  = ssum / (float)KSHOT;
    }
}

// ---------------- raw copy of q into output ----------------
__global__ void copyq_kernel(const float4* __restrict__ q4, float4* __restrict__ out4, int n4) {
    int i = blockIdx.x * blockDim.x + threadIdx.x;
    if (i < n4) out4[i] = q4[i];
}

extern "C" void kernel_launch(void* const* d_in, const int* in_sizes, int n_in,
                              void* d_out, int out_size, void* d_ws, size_t ws_size,
                              hipStream_t stream) {
    const float* s = (const float*)d_in[0];   // support (1024, 768)
    const float* q = (const float*)d_in[1];   // query   (32768, 768)
    float* out = (float*)d_out;

    // output layout (flat, reference return order)
    float* out_proto   = out;                                 // 64*768
    float* out_q       = out_proto + (size_t)NWAY * DIM;      // 32768*768
    float* out_acc     = out_q + (size_t)NQ * DIM;            // 1
    float* out_orig    = out_acc + 1;                         // 64*768
    float* out_sampled = out_orig + (size_t)NWAY * DIM;       // 64*512*768

    // workspace carve-out (256B aligned)
    char* ws = (char*)d_ws;
    size_t off = 0;
    auto take = [&](size_t bytes) -> char* {
        char* p = ws + off;
        off = (off + bytes + 255) & ~(size_t)255;
        return p;
    };
    unsigned short* sbf = (unsigned short*)take((size_t)NS * DIM * 2);
    unsigned short* qbf = (unsigned short*)take((size_t)NQ * DIM * 2);
    float* snorm = (float*)take((size_t)NS * 4);
    float* qnorm = (float*)take((size_t)NQ * 4);
    float* d2    = (float*)take((size_t)NS * NQ * 4);
    float* tval  = (float*)take((size_t)NS * TOPK * 4);
    int*   tidx  = (int*)take((size_t)NS * TOPK * 4);

    // 1) bf16 conversion of both operands
    {
        int n = NS * DIM;
        cvt_bf16_kernel<<<(n + 255) / 256, 256, 0, stream>>>(s, sbf, n);
    }
    {
        int n = NQ * DIM;
        cvt_bf16_kernel<<<(n + 255) / 256, 256, 0, stream>>>(q, qbf, n);
    }
    // 2) exact f32 row norms (one wave per row)
    rownorm_kernel<<<(NS * 32 + 255) / 256, 256, 0, stream>>>(s, snorm, NS);
    rownorm_kernel<<<(NQ * 32 + 255) / 256, 256, 0, stream>>>(q, qnorm, NQ);
    // 3) WMMA distance GEMM: 128x128 tiles, async double-buffered LDS
    {
        dim3 grid(NQ / BN, NS / BM);           // (256, 8)
        dist_gemm_kernel<<<grid, 256, 0, stream>>>(sbf, qbf, snorm, qnorm, d2);
    }
    // 4) top-32 per row (sorted ascending, lowest-index tie-break)
    topk_kernel<<<NS, 256, 0, stream>>>(d2, tval, tidx);
    // 5) accuracy scalar
    acc_kernel<<<1, 256, 0, stream>>>(tidx, out_acc);
    // 6) gather sampled_data
    gather_kernel<<<NS * TOPK, 256, 0, stream>>>(q, tidx, out_sampled);
    // 7) prototypes + original prototypes
    proto_kernel<<<NWAY, 256, 0, stream>>>(s, out_sampled, out_proto, out_orig);
    // 8) raw q copy
    {
        int n4 = NQ * DIM / 4;
        copyq_kernel<<<(n4 + 255) / 256, 256, 0, stream>>>((const float4*)q, (float4*)out_q, n4);
    }
}
